// TripletCenterLossV2_15917148799624
// MI455X (gfx1250) — matrix operand order repaired and checked by reading, not compiled, not run on hardware
//
#include <hip/hip_runtime.h>
#include <hip/hip_bf16.h>
#include <math.h>

#define BATCH    16384
#define NCLS     1000
#define NCLS_PAD 1024
#define FEA      512
#define MARGIN   5.0f
#define BIGF     3.0e38f

typedef __attribute__((ext_vector_type(16))) __bf16 v16bf;
typedef __attribute__((ext_vector_type(8)))  __bf16 v8bf;
typedef __attribute__((ext_vector_type(8)))  float  v8f;
typedef __attribute__((ext_vector_type(4)))  float  v4f;

// ---------------------------------------------------------------------------
// Kernel 1: split centers into bf16 hi/lo (padded to 1024 classes) + ||c||^2
// ---------------------------------------------------------------------------
__global__ __launch_bounds__(128)
void prep_centers(const float* __restrict__ centers,
                  __bf16* __restrict__ chi, __bf16* __restrict__ clo,
                  float* __restrict__ csq) {
  const int c = blockIdx.x;     // 0..1023
  const int t = threadIdx.x;    // 0..127 (one float4 each: 128*4 = 512)
  __shared__ float red[128];
  float s = 0.f;
  if (c < NCLS) {
    const v4f* row = (const v4f*)(centers + (size_t)c * FEA);
    v4f v = row[t];
#pragma unroll
    for (int k = 0; k < 4; ++k) {
      float f = v[k];
      s += f * f;
      __bf16 h = (__bf16)f;                    // RNE hi part
      __bf16 l = (__bf16)(f - (float)h);       // residual lo part
      chi[(size_t)c * FEA + 4 * t + k] = h;
      clo[(size_t)c * FEA + 4 * t + k] = l;
    }
  } else {                                     // zero-fill padded classes
#pragma unroll
    for (int k = 0; k < 4; ++k) {
      chi[(size_t)c * FEA + 4 * t + k] = (__bf16)0.f;
      clo[(size_t)c * FEA + 4 * t + k] = (__bf16)0.f;
    }
  }
  red[t] = s;
  __syncthreads();
  for (int off = 64; off > 0; off >>= 1) {
    if (t < off) red[t] += red[t + off];
    __syncthreads();
  }
  if (t == 0) csq[c] = (c < NCLS) ? red[0] : BIGF;
}

// ---------------------------------------------------------------------------
// Kernel 2: fused distance-GEMM (split-bf16 WMMA) + pos gather + neg min
// One block = 16 rows; 8 waves sweep 16-class column tiles.
// ---------------------------------------------------------------------------
__global__ __launch_bounds__(256)
void tcl_main(const float* __restrict__ x, const int* __restrict__ labels,
              const __bf16* __restrict__ chi, const __bf16* __restrict__ clo,
              const float* __restrict__ csq, float* __restrict__ partial) {
  // Padded row stride 520 halves (1040 B) -> lane m hits bank 4*m, no conflicts
  __shared__ __align__(16) __bf16 shi[16][520];
  __shared__ __align__(16) __bf16 slo[16][520];
  __shared__ float xsq_l[16];
  __shared__ int   lab_l[16];
  __shared__ float pos_l[16];
  __shared__ float negw[8][16];
  __shared__ float term_l[16];

  const int tid  = threadIdx.x;
  const int lane = tid & 31;
  const int wave = tid >> 5;
  const int row0 = blockIdx.x * 16;

  // ---- stage x-strip into LDS as bf16 hi/lo, and compute ||x||^2 per row ----
  {
    const int r   = tid >> 4;   // row 0..15 (16 threads per row)
    const int c16 = tid & 15;
    const v4f* xr = (const v4f*)(x + (size_t)(row0 + r) * FEA);
    float s = 0.f;
#pragma unroll
    for (int i = 0; i < 8; ++i) {
      int j = c16 + i * 16;     // coalesced float4 index 0..127
      v4f v = xr[j];
#pragma unroll
      for (int k = 0; k < 4; ++k) {
        float f = v[k];
        s += f * f;
        __bf16 h = (__bf16)f;
        shi[r][4 * j + k] = h;
        slo[r][4 * j + k] = (__bf16)(f - (float)h);
      }
    }
    // width-16 butterfly sum (threads of one row are one half-wave)
    s += __shfl_xor(s, 1, 16);
    s += __shfl_xor(s, 2, 16);
    s += __shfl_xor(s, 4, 16);
    s += __shfl_xor(s, 8, 16);
    if (c16 == 0) xsq_l[r] = s;
  }
  if (tid < 16) {
    lab_l[tid] = labels[row0 + tid];
    pos_l[tid] = 0.f;
  }
  __syncthreads();

  const int grp   = lane >> 4;   // half-wave: 0 or 1
  const int nlane = lane & 15;
  float regmin[8];
#pragma unroll
  for (int r = 0; r < 8; ++r) regmin[r] = BIGF;

  for (int t = wave; t < NCLS_PAD / 16; t += 8) {
    const int n0 = t * 16;
    const int n  = n0 + nlane;                      // this lane's class column
    const __bf16* bh = chi + (size_t)n * FEA + grp * 16;
    const __bf16* bl = clo + (size_t)n * FEA + grp * 16;
    v8f acc = {};

#pragma unroll
    for (int kk = 0; kk < FEA; kk += 32) {
      // A fragment (k order chosen to match B): lane holds row M=nlane,
      // contiguous K = kk+g..kk+g+15, g = 16*grp
      const int g = grp * 16;
      v8bf ah0 = *(const v8bf*)&shi[nlane][kk + g];
      v8bf ah1 = *(const v8bf*)&shi[nlane][kk + g + 8];
      v8bf al0 = *(const v8bf*)&slo[nlane][kk + g];
      v8bf al1 = *(const v8bf*)&slo[nlane][kk + g + 8];
      v16bf ahi = __builtin_shufflevector(ah0, ah1, 0,1,2,3,4,5,6,7,8,9,10,11,12,13,14,15);
      v16bf alo = __builtin_shufflevector(al0, al1, 0,1,2,3,4,5,6,7,8,9,10,11,12,13,14,15);
      // B fragment: lane holds column N=nlane, contiguous K = kk+16*grp .. +15
      // (B = centers^T, so this is 32 contiguous bytes of one center row)
      v8bf bh0 = *(const v8bf*)(bh + kk);
      v8bf bh1 = *(const v8bf*)(bh + kk + 8);
      v8bf bl0 = *(const v8bf*)(bl + kk);
      v8bf bl1 = *(const v8bf*)(bl + kk + 8);
      v16bf bhi = __builtin_shufflevector(bh0, bh1, 0,1,2,3,4,5,6,7,8,9,10,11,12,13,14,15);
      v16bf blo = __builtin_shufflevector(bl0, bl1, 0,1,2,3,4,5,6,7,8,9,10,11,12,13,14,15);
      // split-bf16: dot = hi*hi + hi*lo + lo*hi (lo*lo term ~2^-16 relative, dropped)
      acc = __builtin_amdgcn_wmma_f32_16x16x32_bf16(false, alo, false, bhi,
                                                    (short)0, acc, false, false);
      acc = __builtin_amdgcn_wmma_f32_16x16x32_bf16(false, ahi, false, blo,
                                                    (short)0, acc, false, false);
      acc = __builtin_amdgcn_wmma_f32_16x16x32_bf16(false, ahi, false, bhi,
                                                    (short)0, acc, false, false);
    }

    // epilogue: C/D layout -> VGPR r holds M = r + 8*grp, N = nlane
    const float cs = csq[n];
#pragma unroll
    for (int r = 0; r < 8; ++r) {
      const int m = r + grp * 8;
      float dist = fmaxf(xsq_l[m] + cs - 2.0f * acc[r], 0.f) * 0.5f;
      const bool ispos = (n == lab_l[m]);
      if (ispos) pos_l[m] = dist;                 // unique (row,label) writer
      float nv = (n < NCLS && !ispos) ? dist : BIGF;
      nv = fminf(nv, __shfl_xor(nv, 1, 16));      // min over 16 classes
      nv = fminf(nv, __shfl_xor(nv, 2, 16));
      nv = fminf(nv, __shfl_xor(nv, 4, 16));
      nv = fminf(nv, __shfl_xor(nv, 8, 16));
      regmin[r] = fminf(regmin[r], nv);
    }
  }

  if (nlane == 0) {
#pragma unroll
    for (int r = 0; r < 8; ++r) negw[wave][r + grp * 8] = regmin[r];
  }
  __syncthreads();

  if (tid < 16) {
    float ng = negw[0][tid];
#pragma unroll
    for (int w = 1; w < 8; ++w) ng = fminf(ng, negw[w][tid]);
    term_l[tid] = log1pf(expf(pos_l[tid] + MARGIN - ng));
  }
  __syncthreads();
  if (tid == 0) {
    float s = 0.f;
#pragma unroll
    for (int i = 0; i < 16; ++i) s += term_l[i];
    partial[blockIdx.x] = s;   // deterministic per-block partial
  }
}

// ---------------------------------------------------------------------------
// Kernel 3: deterministic final reduction -> mean loss
// ---------------------------------------------------------------------------
__global__ __launch_bounds__(256)
void tcl_finalize(const float* __restrict__ partial, float* __restrict__ out) {
  __shared__ float red[256];
  const int t = threadIdx.x;
  float s = 0.f;
  for (int i = t; i < BATCH / 16; i += 256) s += partial[i];
  red[t] = s;
  __syncthreads();
  for (int off = 128; off > 0; off >>= 1) {
    if (t < off) red[t] += red[t + off];
    __syncthreads();
  }
  if (t == 0) out[0] = red[0] / (float)BATCH;
}

// ---------------------------------------------------------------------------
extern "C" void kernel_launch(void* const* d_in, const int* in_sizes, int n_in,
                              void* d_out, int out_size, void* d_ws, size_t ws_size,
                              hipStream_t stream) {
  const float* x       = (const float*)d_in[0];
  const float* centers = (const float*)d_in[1];
  const int*   labels  = (const int*)d_in[2];

  // workspace layout
  char*   ws      = (char*)d_ws;
  __bf16* chi     = (__bf16*)ws;                               // 1024*512 bf16
  __bf16* clo     = chi + (size_t)NCLS_PAD * FEA;              // 1024*512 bf16
  float*  csq     = (float*)(clo + (size_t)NCLS_PAD * FEA);    // 1024 f32
  float*  partial = csq + NCLS_PAD;                            // 1024 f32

  prep_centers<<<NCLS_PAD, 128, 0, stream>>>(centers, chi, clo, csq);
  tcl_main<<<BATCH / 16, 256, 0, stream>>>(x, labels, chi, clo, csq, partial);
  tcl_finalize<<<1, 256, 0, stream>>>(partial, (float*)d_out);
}
